// AttentionEncoder_24369644438113
// MI455X (gfx1250) — compile-verified, matrix operand order checked by reading, not run on hardware
//
#include <hip/hip_runtime.h>

// ---------------------------------------------------------------------------
// Problem constants
// ---------------------------------------------------------------------------
#define Bb   32
#define Ll   577
#define Dd   512
#define Hh   8
#define DK   64
#define FFd  2048
#define BL   (Bb*Ll)            // 18464
#define LND  (Ll*Dd)            // 295424
#define QT   37                 // ceil(577/16)

typedef __bf16 bf16_t;
typedef __attribute__((ext_vector_type(16))) __bf16 v16bf;
typedef __attribute__((ext_vector_type(8)))  float  v8f;

union FragU {
    v16bf  v;
    uint4  q[2];
    bf16_t e[16];
};

__device__ __forceinline__ uint4 ld16g(const bf16_t* p) {
    return *reinterpret_cast<const uint4*>(p);
}
__device__ __forceinline__ void zero_frag(FragU& f) {
    f.q[0] = make_uint4(0u,0u,0u,0u);
    f.q[1] = make_uint4(0u,0u,0u,0u);
}
__device__ __forceinline__ v8f wmma_bf16(v16bf a, v16bf b, v8f c) {
    return __builtin_amdgcn_wmma_f32_16x16x32_bf16(
        /*neg_a=*/false, a, /*neg_b=*/false, b,
        /*c_mod=*/(short)0, c, /*reuse_a=*/false, /*reuse_b=*/false);
}

// Async global -> LDS copy, 16 bytes per lane (gfx1250, ASYNCcnt tracked).
// LDS aperture keeps the byte offset in flat-addr[31:0], so truncating the
// generic shared pointer yields the VDST LDS address.
__device__ __forceinline__ void async_cp16(void* smem, const void* gmem) {
    unsigned int loff = (unsigned int)(size_t)smem;
    asm volatile("global_load_async_to_lds_b128 %0, %1, off"
                 :: "v"(loff), "v"(gmem) : "memory");
}
__device__ __forceinline__ void wait_async0() {
    asm volatile("s_wait_asynccnt 0x0" ::: "memory");
}

// ---------------------------------------------------------------------------
// Weight convert + transpose: wt[n*K + k] = (bf16) w[k*N + n]
// ---------------------------------------------------------------------------
__global__ void cvt_wt_kernel(const float* __restrict__ w, bf16_t* __restrict__ wt,
                              int K, int N) {
    int idx = blockIdx.x * blockDim.x + threadIdx.x;
    int total = K * N;
    if (idx < total) {
        int n = idx / K;
        int k = idx - n * K;
        wt[idx] = (bf16_t)w[(size_t)k * N + n];
    }
}

// ---------------------------------------------------------------------------
// Whole-sequence LayerNorm over L*D elements per batch, unbiased std,
// out = gamma*((x-mean)/(std+eps)) + beta, written as bf16.
// ---------------------------------------------------------------------------
__global__ __launch_bounds__(512) void ln_kernel(const float* __restrict__ x,
                                                 const float* __restrict__ gamma,
                                                 const float* __restrict__ beta,
                                                 bf16_t* __restrict__ out) {
    const int b = blockIdx.x;
    const float* xb = x + (size_t)b * LND;
    bf16_t* ob = out + (size_t)b * LND;

    float s = 0.f, s2 = 0.f;
    for (int i = threadIdx.x; i < LND; i += blockDim.x) {
        float v = xb[i];
        s += v; s2 += v * v;
    }
    __shared__ float red[512];
    __shared__ float red2[512];
    red[threadIdx.x] = s; red2[threadIdx.x] = s2;
    __syncthreads();
    for (int st = 256; st > 0; st >>= 1) {
        if ((int)threadIdx.x < st) {
            red[threadIdx.x]  += red[threadIdx.x + st];
            red2[threadIdx.x] += red2[threadIdx.x + st];
        }
        __syncthreads();
    }
    float S = red[0], S2 = red2[0];
    float mean = S / (float)LND;
    float var  = (S2 - S * mean) / (float)(LND - 1);
    float inv  = 1.f / (sqrtf(fmaxf(var, 0.f)) + 1e-6f);
    for (int i = threadIdx.x; i < LND; i += blockDim.x) {
        float v = (xb[i] - mean) * inv;
        ob[i] = (bf16_t)(gamma[i] * v + beta[i]);
    }
}

// ---------------------------------------------------------------------------
// WMMA GEMM with double-buffered async global->LDS staging.
// Block = 256 thr = 8 waves. Block tile 128x128, K step 32.
// Wave tile 32x64 (2 M-frags x 4 N-frags). A[M,K] bf16, WT[N,K] bf16.
// M edge handled by clamping staged rows; N must be a multiple of 128.
// ---------------------------------------------------------------------------
#define LDS_ROW 40   // bf16 elements per LDS row (32 data + 8 pad = 80 B)

template<bool GELU, bool RES, bool OUTBF>
__global__ __launch_bounds__(256) void gemm_kernel(const bf16_t* __restrict__ A,
                                                   const bf16_t* __restrict__ WT,
                                                   const float* __restrict__ bias,
                                                   const float* res,
                                                   float* outF, bf16_t* outB,
                                                   int M, int N, int K) {
    __shared__ bf16_t sA[2][128 * LDS_ROW];
    __shared__ bf16_t sB[2][128 * LDS_ROW];

    const int tid  = threadIdx.x;
    const int lane = tid & 31;
    const int wave = tid >> 5;
    const int lr   = lane & 15;
    const int kb   = (lane >> 4) * 8;
    const int wm   = wave >> 1;            // 0..3  (32-row slice)
    const int wn   = wave & 1;             // 0..1  (64-col slice)
    const int bm0  = blockIdx.y * 128;
    const int bn0  = blockIdx.x * 128;

    // Stage one 128x32 A tile and one 128x32 B tile (each 128 rows x 64 B).
    // 512 16-byte chunks per matrix; each thread copies 2 chunks of each.
    auto stage = [&](int buf, int kk) {
        #pragma unroll
        for (int r = 0; r < 2; ++r) {
            int c   = tid + r * 256;       // 0..511
            int row = c >> 2;              // 0..127
            int bof = (c & 3) * 16;        // byte offset within 64B row
            int arow = bm0 + row; if (arow > M - 1) arow = M - 1;
            const char* ga = (const char*)(A + (size_t)arow * K + kk) + bof;
            async_cp16((char*)&sA[buf][0] + row * (LDS_ROW * 2) + bof, ga);
            const char* gb = (const char*)(WT + (size_t)(bn0 + row) * K + kk) + bof;
            async_cp16((char*)&sB[buf][0] + row * (LDS_ROW * 2) + bof, gb);
        }
    };

    v8f acc[2][4] = {};
    const int steps = K / 32;
    stage(0, 0);

    for (int s = 0; s < steps; ++s) {
        int cur = s & 1;
        wait_async0();
        __syncthreads();
        if (s + 1 < steps) stage(cur ^ 1, (s + 1) * 32);

        FragU af[2], bfm[4];
        #pragma unroll
        for (int mi = 0; mi < 2; ++mi) {
            const bf16_t* p = &sA[cur][(wm * 32 + 16 * mi + lr) * LDS_ROW + kb];
            af[mi].q[0] = *reinterpret_cast<const uint4*>(p);
            af[mi].q[1] = *reinterpret_cast<const uint4*>(p + 16);
        }
        #pragma unroll
        for (int ni = 0; ni < 4; ++ni) {
            const bf16_t* p = &sB[cur][(wn * 64 + 16 * ni + lr) * LDS_ROW + kb];
            bfm[ni].q[0] = *reinterpret_cast<const uint4*>(p);
            bfm[ni].q[1] = *reinterpret_cast<const uint4*>(p + 16);
        }
        #pragma unroll
        for (int mi = 0; mi < 2; ++mi)
            #pragma unroll
            for (int ni = 0; ni < 4; ++ni)
                acc[mi][ni] = wmma_bf16(af[mi].v, bfm[ni].v, acc[mi][ni]);
        __syncthreads();
    }

    // Epilogue
    #pragma unroll
    for (int mi = 0; mi < 2; ++mi) {
        int rb = bm0 + wm * 32 + 16 * mi + (lane >> 4) * 8;
        #pragma unroll
        for (int ni = 0; ni < 4; ++ni) {
            int col = bn0 + wn * 64 + 16 * ni + lr;
            float bv = bias[col];
            #pragma unroll
            for (int i = 0; i < 8; ++i) {
                int row = rb + i;
                if (row < M) {
                    float v = acc[mi][ni][i] + bv;
                    if (GELU) v = 0.5f * v * (1.f + erff(v * 0.7071067811865476f));
                    if (RES)  v += res[(size_t)row * N + col];
                    if (OUTBF) outB[(size_t)row * N + col] = (bf16_t)v;
                    else       outF[(size_t)row * N + col] = v;
                }
            }
        }
    }
}

// ---------------------------------------------------------------------------
// Flash attention: one wave per (b, h, 16-query-row tile). Key blocks of 32.
// Q/K/V are bf16 in [B*L, H*DK] layout. Output O in same layout (bf16).
// ---------------------------------------------------------------------------
__global__ __launch_bounds__(256) void attn_kernel(const bf16_t* __restrict__ Q,
                                                   const bf16_t* __restrict__ Km,
                                                   const bf16_t* __restrict__ V,
                                                   bf16_t* __restrict__ O) {
    __shared__ bf16_t pls[8 * 16 * 32];   // per-wave 16x32 P tile
    const int lane = threadIdx.x & 31;
    const int wv   = threadIdx.x >> 5;
    const int wid  = blockIdx.x * 8 + wv;
    const int qt   = wid % QT;
    const int h    = (wid / QT) & (Hh - 1);
    const int b    = wid / (QT * Hh);
    const int lr   = lane & 15;
    const int kb   = (lane >> 4) * 8;
    const int rhalf = (lane >> 4) * 8;    // row base within 16-row tile for C frags

    const bf16_t* qp = Q + ((size_t)b * Ll) * Dd + h * DK;
    const bf16_t* kp = Km + ((size_t)b * Ll) * Dd + h * DK;
    const bf16_t* vp = V + ((size_t)b * Ll) * Dd + h * DK;
    bf16_t* lds = pls + wv * 512;

    // Load the 16x64 Q tile as two A-frags (K-steps of 32)
    FragU aq[2];
    {
        int row = qt * 16 + lr;
        if (row < Ll) {
            const bf16_t* p = qp + (size_t)row * Dd;
            #pragma unroll
            for (int s = 0; s < 2; ++s) {
                aq[s].q[0] = ld16g(p + s * 32 + kb);
                aq[s].q[1] = ld16g(p + s * 32 + kb + 16);
            }
        } else {
            zero_frag(aq[0]); zero_frag(aq[1]);
        }
    }

    v8f   oacc[4] = {};
    float mrun[8], lrun[8];
    #pragma unroll
    for (int i = 0; i < 8; ++i) { mrun[i] = -3.0e38f; lrun[i] = 0.f; }
    const float scale = 0.044194173824159216f;  // 1/sqrt(512)

    for (int mb = 0; mb < 608; mb += 32) {
        // ---- S = Q K^T for 16x32 key block (two 16x16 C frags) ----
        v8f sfr[2] = {};
        #pragma unroll
        for (int j = 0; j < 2; ++j) {
            int key = mb + 16 * j + lr;
            FragU bk[2];
            if (key < Ll) {
                const bf16_t* p = kp + (size_t)key * Dd;
                #pragma unroll
                for (int s = 0; s < 2; ++s) {
                    bk[s].q[0] = ld16g(p + s * 32 + kb);
                    bk[s].q[1] = ld16g(p + s * 32 + kb + 16);
                }
            } else {
                zero_frag(bk[0]); zero_frag(bk[1]);
            }
            sfr[j] = wmma_bf16(aq[0].v, bk[0].v, sfr[j]);
            sfr[j] = wmma_bf16(aq[1].v, bk[1].v, sfr[j]);
            bool oob = (key >= Ll);
            #pragma unroll
            for (int i = 0; i < 8; ++i)
                sfr[j][i] = oob ? -3.0e38f : sfr[j][i] * scale;
        }

        // ---- row max over 32 columns ----
        float rmax[8];
        #pragma unroll
        for (int i = 0; i < 8; ++i) rmax[i] = fmaxf(sfr[0][i], sfr[1][i]);
        #pragma unroll
        for (int m = 1; m < 16; m <<= 1)
            #pragma unroll
            for (int i = 0; i < 8; ++i)
                rmax[i] = fmaxf(rmax[i], __shfl_xor(rmax[i], m, 32));

        // ---- online softmax update ----
        float p0[8], p1[8], alp[8], rs[8];
        #pragma unroll
        for (int i = 0; i < 8; ++i) {
            float mn = fmaxf(mrun[i], rmax[i]);
            alp[i] = __expf(mrun[i] - mn);
            p0[i]  = __expf(sfr[0][i] - mn);
            p1[i]  = __expf(sfr[1][i] - mn);
            mrun[i] = mn;
            rs[i] = p0[i] + p1[i];
        }
        #pragma unroll
        for (int m = 1; m < 16; m <<= 1)
            #pragma unroll
            for (int i = 0; i < 8; ++i)
                rs[i] += __shfl_xor(rs[i], m, 32);
        #pragma unroll
        for (int i = 0; i < 8; ++i) lrun[i] = lrun[i] * alp[i] + rs[i];
        #pragma unroll
        for (int t = 0; t < 4; ++t)
            #pragma unroll
            for (int i = 0; i < 8; ++i)
                oacc[t][i] *= alp[i];

        // ---- C-layout P -> LDS -> A-layout bf16 frag ----
        #pragma unroll
        for (int j = 0; j < 2; ++j) {
            int colb = 16 * j + lr;
            #pragma unroll
            for (int i = 0; i < 8; ++i)
                lds[(rhalf + i) * 32 + colb] = (bf16_t)(j ? p1[i] : p0[i]);
        }
        __builtin_amdgcn_wave_barrier();  // keep scheduler from crossing the bounce
        FragU ap;
        {
            const bf16_t* lp = lds + lr * 32 + kb;
            ap.q[0] = *reinterpret_cast<const uint4*>(lp);
            ap.q[1] = *reinterpret_cast<const uint4*>(lp + 16);
        }
        __builtin_amdgcn_wave_barrier();

        // ---- O += P @ V (four 16x16 N tiles, K=32) ----
        #pragma unroll
        for (int t = 0; t < 4; ++t) {
            FragU bv;
            int c = 16 * t + lr;
            #pragma unroll
            for (int i = 0; i < 8; ++i) {
                int tok0 = mb + kb + i;
                int tok1 = mb + kb + 16 + i;
                bv.e[i]     = (tok0 < Ll) ? vp[(size_t)tok0 * Dd + c] : (bf16_t)0.0f;
                bv.e[i + 8] = (tok1 < Ll) ? vp[(size_t)tok1 * Dd + c] : (bf16_t)0.0f;
            }
            oacc[t] = wmma_bf16(ap.v, bv.v, oacc[t]);
        }
    }

    // ---- write O / l ----
    #pragma unroll
    for (int t = 0; t < 4; ++t) {
        int col = h * DK + 16 * t + lr;
        int rowb = qt * 16 + rhalf;
        #pragma unroll
        for (int i = 0; i < 8; ++i) {
            int row = rowb + i;
            if (row < Ll)
                O[((size_t)b * Ll + row) * Dd + col] = (bf16_t)(oacc[t][i] / lrun[i]);
        }
    }
}

// ---------------------------------------------------------------------------
// Host launcher
// ---------------------------------------------------------------------------
extern "C" void kernel_launch(void* const* d_in, const int* in_sizes, int n_in,
                              void* d_out, int out_size, void* d_ws, size_t ws_size,
                              hipStream_t stream) {
    const float* x      = (const float*)d_in[0];
    const float* gamma1 = (const float*)d_in[1];
    const float* beta1  = (const float*)d_in[2];
    const float* gamma2 = (const float*)d_in[3];
    const float* beta2  = (const float*)d_in[4];
    const float* wq = (const float*)d_in[5];
    const float* bq = (const float*)d_in[6];
    const float* wk = (const float*)d_in[7];
    const float* bk = (const float*)d_in[8];
    const float* wv = (const float*)d_in[9];
    const float* bv = (const float*)d_in[10];
    const float* wo = (const float*)d_in[11];
    const float* bo = (const float*)d_in[12];
    const float* w1 = (const float*)d_in[13];
    const float* b1 = (const float*)d_in[14];
    const float* w2 = (const float*)d_in[15];
    const float* b2 = (const float*)d_in[16];

    char* ws = (char*)d_ws;
    const size_t SZ_W512  = (size_t)Dd * Dd * 2;          //   512 KiB
    const size_t SZ_WFF   = (size_t)Dd * FFd * 2;         //     2 MiB
    const size_t SZ_ACT   = (size_t)BL * Dd * 2;          //  ~18 MiB (bf16)

    bf16_t* wqT = (bf16_t*)(ws);
    bf16_t* wkT = (bf16_t*)(ws + SZ_W512);
    bf16_t* wvT = (bf16_t*)(ws + 2 * SZ_W512);
    bf16_t* woT = (bf16_t*)(ws + 3 * SZ_W512);
    bf16_t* w1T = (bf16_t*)(ws + 4 * SZ_W512);
    bf16_t* w2T = (bf16_t*)(ws + 4 * SZ_W512 + SZ_WFF);
    size_t off = 4 * SZ_W512 + 2 * SZ_WFF;                // 6 MiB
    bf16_t* h1b = (bf16_t*)(ws + off);                    // LN1 out; reused for LN2 out
    bf16_t* qb  = (bf16_t*)(ws + off + SZ_ACT);
    bf16_t* kbf = (bf16_t*)(ws + off + 2 * SZ_ACT);
    bf16_t* vb  = (bf16_t*)(ws + off + 3 * SZ_ACT);
    bf16_t* ob  = (bf16_t*)(ws + off + 4 * SZ_ACT);
    bf16_t* fb  = (bf16_t*)(ws + off + SZ_ACT);           // FFN1 out aliases q/k/v/o (4*SZ_ACT == SZ_FF)
    float*  hres = (float*)d_out;                          // residual-1 output lives in d_out

    // 1) weights -> bf16 transposed [N,K]
    cvt_wt_kernel<<<(Dd * Dd + 255) / 256, 256, 0, stream>>>(wq, wqT, Dd, Dd);
    cvt_wt_kernel<<<(Dd * Dd + 255) / 256, 256, 0, stream>>>(wk, wkT, Dd, Dd);
    cvt_wt_kernel<<<(Dd * Dd + 255) / 256, 256, 0, stream>>>(wv, wvT, Dd, Dd);
    cvt_wt_kernel<<<(Dd * Dd + 255) / 256, 256, 0, stream>>>(wo, woT, Dd, Dd);
    cvt_wt_kernel<<<(Dd * FFd + 255) / 256, 256, 0, stream>>>(w1, w1T, Dd, FFd);
    cvt_wt_kernel<<<(FFd * Dd + 255) / 256, 256, 0, stream>>>(w2, w2T, FFd, Dd);

    // 2) LN1
    ln_kernel<<<Bb, 512, 0, stream>>>(x, gamma1, beta1, h1b);

    // 3) Q/K/V projections  (block tile 128x128)
    dim3 g512(Dd / 128, (BL + 127) / 128);    // (4, 145)
    gemm_kernel<false,false,true><<<g512, 256, 0, stream>>>(h1b, wqT, bq, nullptr, nullptr, qb, BL, Dd, Dd);
    gemm_kernel<false,false,true><<<g512, 256, 0, stream>>>(h1b, wkT, bk, nullptr, nullptr, kbf, BL, Dd, Dd);
    gemm_kernel<false,false,true><<<g512, 256, 0, stream>>>(h1b, wvT, bv, nullptr, nullptr, vb, BL, Dd, Dd);

    // 4) attention (B*H*QT = 9472 waves = 1184 blocks of 8 waves)
    attn_kernel<<<(Bb * Hh * QT) / 8, 256, 0, stream>>>(qb, kbf, vb, ob);

    // 5) O projection + residual(x) -> h (f32 in d_out)
    gemm_kernel<false,true,false><<<g512, 256, 0, stream>>>(ob, woT, bo, x, hres, nullptr, BL, Dd, Dd);

    // 6) LN2 -> g (bf16, reuse h1 buffer)
    ln_kernel<<<Bb, 512, 0, stream>>>(hres, gamma2, beta2, h1b);

    // 7) FFN1 + exact GELU -> fb (bf16)
    dim3 gff(FFd / 128, (BL + 127) / 128);    // (16, 145)
    gemm_kernel<true,false,true><<<gff, 256, 0, stream>>>(h1b, w1T, b1, nullptr, nullptr, fb, BL, FFd, Dd);

    // 8) FFN2 + residual(h) -> d_out (reads and writes same element per thread)
    gemm_kernel<false,true,false><<<g512, 256, 0, stream>>>(fb, w2T, b2, hres, hres, nullptr, BL, Dd, FFd);
}
